// SMoE_56324201120511
// MI455X (gfx1250) — matrix operand order repaired and checked by reading, not compiled, not run on hardware
//
#include <hip/hip_runtime.h>

// ---- problem constants (fixed by reference) ----
#define TT   8192      // tokens = 4*2048
#define DD   1024      // model dim
#define HH   4096      // expert hidden
#define EE   8         // experts
#define CC   1280      // capacity = T/E * 1.25

// ---- vector typedefs (clang ext_vector: POD, union-safe) ----
typedef __attribute__((ext_vector_type(16))) __bf16        bf16x16;
typedef __attribute__((ext_vector_type(8)))  float         floatx8;
typedef __attribute__((ext_vector_type(4)))  float         fx4;
typedef __attribute__((ext_vector_type(4)))  unsigned int  ux4;
typedef __attribute__((ext_vector_type(2)))  unsigned int  ux2;
typedef __attribute__((ext_vector_type(4)))  unsigned int  u32x4;
typedef __attribute__((ext_vector_type(4)))  int           i32x4;
typedef __attribute__((ext_vector_type(8)))  int           i32x8;

#if defined(__has_builtin)
#if __has_builtin(__builtin_amdgcn_tensor_load_to_lds) && \
    __has_builtin(__builtin_amdgcn_s_wait_tensorcnt)
#define HAVE_TDM 1
#endif
#endif
#ifndef HAVE_TDM
#define HAVE_TDM 0
#endif

__device__ __forceinline__ unsigned short f2bf(float f) {
  unsigned u = __builtin_bit_cast(unsigned, f);
  unsigned r = u + 0x7FFFu + ((u >> 16) & 1u);   // round-to-nearest-even
  return (unsigned short)(r >> 16);
}

// load a 16-element bf16 WMMA fragment as two contiguous b128 chunks
__device__ __forceinline__ bf16x16 ld_frag(const unsigned short* p0,
                                           const unsigned short* p1) {
  union { bf16x16 v; ux4 q[2]; } f;
  f.q[0] = *(const ux4*)p0;
  f.q[1] = *(const ux4*)p1;
  return f.v;
}

__device__ __forceinline__ floatx8 zero8() {
  floatx8 z = {0.f,0.f,0.f,0.f,0.f,0.f,0.f,0.f};
  return z;
}

#if HAVE_TDM
// ---------------------------------------------------------------
// Tensor Data Mover: async-DMA a 2-D bf16 tile (tile_rows x tile_k,
// row stride row_stride elements) from global into LDS (packed
// row-major [tile_rows][tile_k]). D# bitfields per CDNA5 ISA ch.8.
// Tracked by TENSORcnt; caller must s_wait_tensorcnt + barrier.
// ---------------------------------------------------------------
__device__ __forceinline__ void tdm_load_2d(unsigned lds_off_bytes,
                                            const void* gptr,
                                            unsigned tile_k,
                                            unsigned tile_rows,
                                            unsigned row_stride) {
  unsigned long long ga = (unsigned long long)gptr;
  u32x4 g0;
  g0.x = 1u;                                         // count=1 (valid), user mode
  g0.y = lds_off_bytes;                              // lds_addr
  g0.z = (unsigned)(ga & 0xFFFFFFFFu);               // global_addr[31:0]
  g0.w = (unsigned)((ga >> 32) & 0x01FFFFFFu)        // global_addr[56:32]
         | (2u << 30);                               // type=2 ("image")
  const unsigned td0 = row_stride;                   // tensor_dim0 (OOB bound)
  const unsigned td1 = 0x00100000u;                  // tensor_dim1: large
  const unsigned long long s0 = (unsigned long long)row_stride;
  i32x8 g1;
  g1[0] = (int)(1u << 16);                           // data_size=1 (2 bytes)
  g1[1] = (int)((td0 & 0xFFFFu) << 16);              // abar=0 | td0[15:0]
  g1[2] = (int)((td0 >> 16) | ((td1 & 0xFFFFu) << 16));
  g1[3] = (int)((td1 >> 16) | (tile_k << 16));       // tile_dim0
  g1[4] = (int)(tile_rows & 0xFFFFu);                // tile_dim1 (tile_dim2=0)
  g1[5] = (int)(s0 & 0xFFFFFFFFu);                   // tensor_dim0_stride[31:0]
  g1[6] = (int)((s0 >> 32) & 0xFFFFu);               // stride[47:32] | d1s[15:0]=0
  g1[7] = 0;
  i32x4 z4 = {0, 0, 0, 0};
#if __clang_major__ >= 23
  i32x8 z8 = {0, 0, 0, 0, 0, 0, 0, 0};
  __builtin_amdgcn_tensor_load_to_lds(g0, g1, z4, z4, z8, 0);
#else
  __builtin_amdgcn_tensor_load_to_lds(g0, g1, z4, z4, 0);
#endif
}
#endif

// ============================================================
// 1) Gating: one wave per token -> top-2 experts + softmax weights
// ============================================================
__global__ __launch_bounds__(256) void moe_gate(
    const float* __restrict__ x, const float* __restrict__ Wg,
    const float* __restrict__ bg,
    int* __restrict__ topi /*[T,2]*/, float* __restrict__ topw /*[T,2]*/) {
  const int t    = (blockIdx.x * 256 + threadIdx.x) >> 5;
  const int lane = threadIdx.x & 31;
  const float* px = x + (size_t)t * DD;

  float acc[EE];
  #pragma unroll
  for (int e = 0; e < EE; ++e) acc[e] = 0.f;

  for (int d0 = 0; d0 < DD; d0 += 32) {
    float xv = px[d0 + lane];
    const float* pw = Wg + (size_t)(d0 + lane) * EE;
    #pragma unroll
    for (int e = 0; e < EE; ++e) acc[e] += xv * pw[e];
  }
  #pragma unroll
  for (int e = 0; e < EE; ++e) {
    #pragma unroll
    for (int off = 16; off > 0; off >>= 1)
      acc[e] += __shfl_down(acc[e], off, 32);
  }
  if (lane == 0) {
    float lg[EE];
    #pragma unroll
    for (int e = 0; e < EE; ++e) lg[e] = acc[e] + bg[e];
    int e0 = 0;
    #pragma unroll
    for (int e = 1; e < EE; ++e) if (lg[e] > lg[e0]) e0 = e;
    int e1 = (e0 == 0) ? 1 : 0;
    #pragma unroll
    for (int e = 0; e < EE; ++e) if (e != e0 && lg[e] > lg[e1]) e1 = e;
    float w0 = 1.f / (1.f + __expf(lg[e1] - lg[e0]));   // softmax over top-2
    topi[t * 2 + 0] = e0;  topi[t * 2 + 1] = e1;
    topw[t * 2 + 0] = w0;  topw[t * 2 + 1] = 1.f - w0;
  }
}

// ============================================================
// 2) Deterministic ordered slot assignment (cumsum semantics).
//    One block; wave e scans all tokens for expert e with ballot/popc.
// ============================================================
__global__ __launch_bounds__(256) void moe_scan(
    const int* __restrict__ topi,
    int* __restrict__ slot_of /*[T,2]*/, int* __restrict__ tok_of /*[E*C]*/) {
  const int lane = threadIdx.x & 31;
  const int e    = threadIdx.x >> 5;
  int cnt = 0;
  for (int base = 0; base < TT; base += 32) {
    const int t  = base + lane;
    const int i0 = topi[t * 2 + 0];
    const int i1 = topi[t * 2 + 1];
    const int k  = (i0 == e) ? 0 : ((i1 == e) ? 1 : -1);
    const bool in = (k >= 0);
    unsigned mask = (unsigned)__ballot(in);
    int slot = cnt + __popc(mask & ((1u << lane) - 1u));
    if (in) {
      if (slot < CC) {
        tok_of[e * CC + slot] = t;
        slot_of[t * 2 + k]    = e * CC + slot;
      } else {
        slot_of[t * 2 + k]    = -1;       // dropped (over capacity)
      }
    }
    cnt += __popc(mask);
  }
  for (int s = cnt + lane; s < CC; s += 32) tok_of[e * CC + s] = -1;
}

// ============================================================
// 3) Dispatch: gather tokens into [E*C, D] bf16 buffers (zero padding)
// ============================================================
__global__ __launch_bounds__(256) void moe_dispatch(
    const float* __restrict__ x, const int* __restrict__ tok_of,
    unsigned short* __restrict__ disp) {
  const int row = blockIdx.x;
  const int t   = tok_of[row];
  const int d   = threadIdx.x * 4;
  unsigned lo = 0u, hi = 0u;
  if (t >= 0) {
    fx4 xv = *(const fx4*)(x + (size_t)t * DD + d);
    lo = (unsigned)f2bf(xv.x) | ((unsigned)f2bf(xv.y) << 16);
    hi = (unsigned)f2bf(xv.z) | ((unsigned)f2bf(xv.w) << 16);
  }
  ux2 v; v.x = lo; v.y = hi;
  *(ux2*)(disp + (size_t)row * DD + d) = v;
}

// ============================================================
// 4) Fused SwiGLU dual-GEMM: h = silu(A@W1) * (A@W2)
//    Block tile 128(M) x 64(N) x 32(K); 8 waves, each 32x32.
//    A tile: TDM async DMA (bf16, no conversion needed).
//    B tiles: fp32->bf16 in-register, packed b32 stores, [n][k] LDS.
// ============================================================
__global__ __launch_bounds__(256) void moe_gemm1(
    const unsigned short* __restrict__ disp, const float* __restrict__ W1,
    const float* __restrict__ W2, unsigned short* __restrict__ hbuf) {
  const int e  = blockIdx.z;
  const int mB = blockIdx.y * 128;
  const int nB = blockIdx.x * 64;

  __shared__ __align__(16) unsigned short lA [128 * 32];  // [m][k]
  __shared__ __align__(16) unsigned short lB1[ 64 * 32];  // [n][k] (transposed)
  __shared__ __align__(16) unsigned short lB2[ 64 * 32];

  const int tid  = threadIdx.x;
  const int lane = tid & 31;
  const int w    = tid >> 5;
  const int wm   = (w & 3) * 32;       // wave M origin within block tile
  const int wn   = (w >> 2) * 32;      // wave N origin

  floatx8 acc1[2][2], acc2[2][2];
  #pragma unroll
  for (int i = 0; i < 2; ++i)
    #pragma unroll
    for (int j = 0; j < 2; ++j) { acc1[i][j] = zero8(); acc2[i][j] = zero8(); }

  const size_t wOff = (size_t)e * DD * HH;
  const float* pW1 = W1 + wOff;
  const float* pW2 = W2 + wOff;
  const unsigned short* pA = disp + (size_t)(e * CC + mB) * DD;
  const unsigned ldsA = (unsigned)(size_t)(void*)lA;

  const int rowL = lane & 15;
  const int c0   = (lane < 16) ? 0 : 8;    // A: K 0-7/16-23 vs 8-15/24-31
  const int kcol = (lane < 16) ? 0 : 8;    // B: K 0-7/16-23 vs 8-15/24-31
  const int nn4  = (tid & 15) * 4;         // B staging: 4 n-cols
  const int kk2  = (tid >> 4) * 2;         // B staging: 2 k-rows

  for (int k0 = 0; k0 < DD; k0 += 32) {
#if HAVE_TDM
    if (w == 0) tdm_load_2d(ldsA, pA + k0, 32, 128, DD);   // async A tile
#else
    {
      const int r = tid >> 1, seg = (tid & 1) * 16;
      const ux4* src = (const ux4*)(pA + (size_t)r * DD + k0 + seg);
      ux4* dst = (ux4*)(lA + r * 32 + seg);
      dst[0] = src[0];
      dst[1] = src[1];
    }
#endif
    { // B tiles: 2 k-rows x 4 n-cols each; pack (k,k+1) pairs -> b32 stores
      const float* r10 = pW1 + (size_t)(k0 + kk2) * HH + nB + nn4;
      const float* r20 = pW2 + (size_t)(k0 + kk2) * HH + nB + nn4;
      if (k0 + 32 < DD) {                         // hint next K-slab into L2
        __builtin_prefetch(r10 + 32 * HH, 0, 1);
        __builtin_prefetch(r20 + 32 * HH, 0, 1);
      }
      fx4 a0 = *(const fx4*)r10;
      fx4 a1 = *(const fx4*)(r10 + HH);
      fx4 b0 = *(const fx4*)r20;
      fx4 b1 = *(const fx4*)(r20 + HH);
      #pragma unroll
      for (int j = 0; j < 4; ++j) {
        *(unsigned*)&lB1[(nn4 + j) * 32 + kk2] =
            (unsigned)f2bf(a0[j]) | ((unsigned)f2bf(a1[j]) << 16);
        *(unsigned*)&lB2[(nn4 + j) * 32 + kk2] =
            (unsigned)f2bf(b0[j]) | ((unsigned)f2bf(b1[j]) << 16);
      }
    }
#if HAVE_TDM
    if (w == 0) __builtin_amdgcn_s_wait_tensorcnt(0);
#endif
    __syncthreads();

    bf16x16 aF[2], b1F[2], b2F[2];
    #pragma unroll
    for (int mt = 0; mt < 2; ++mt) {
      const unsigned short* b = lA + (wm + mt * 16 + rowL) * 32 + c0;
      aF[mt] = ld_frag(b, b + 16);
    }
    #pragma unroll
    for (int nt = 0; nt < 2; ++nt) {
      const unsigned short* b1 = lB1 + (wn + nt * 16 + rowL) * 32 + kcol;
      const unsigned short* b2 = lB2 + (wn + nt * 16 + rowL) * 32 + kcol;
      b1F[nt] = ld_frag(b1, b1 + 16);
      b2F[nt] = ld_frag(b2, b2 + 16);
    }
    #pragma unroll
    for (int mt = 0; mt < 2; ++mt)
      #pragma unroll
      for (int nt = 0; nt < 2; ++nt) {
        acc1[mt][nt] = __builtin_amdgcn_wmma_f32_16x16x32_bf16(
            false, aF[mt], false, b1F[nt], (short)0, acc1[mt][nt], false, false);
        acc2[mt][nt] = __builtin_amdgcn_wmma_f32_16x16x32_bf16(
            false, aF[mt], false, b2F[nt], (short)0, acc2[mt][nt], false, false);
      }
    __syncthreads();
  }

  // epilogue: h = silu(s1) * s2 = s1 * rcp(1+exp(-s1)) * s2, store bf16
  unsigned short* pH = hbuf + (size_t)(e * CC + mB) * HH + nB;
  const int nCol = lane & 15;
  const int rAdd = (lane < 16) ? 0 : 8;
  #pragma unroll
  for (int mt = 0; mt < 2; ++mt)
    #pragma unroll
    for (int nt = 0; nt < 2; ++nt)
      #pragma unroll
      for (int r = 0; r < 8; ++r) {
        const int m = wm + mt * 16 + rAdd + r;
        const int n = wn + nt * 16 + nCol;
        const float s1 = acc1[mt][nt][r];
        const float s2 = acc2[mt][nt][r];
        const float hv = s1 * __builtin_amdgcn_rcpf(1.f + __expf(-s1)) * s2;
        pH[(size_t)m * HH + n] = f2bf(hv);
      }
}

// ============================================================
// 5) out = h @ W3   (K = 4096), fp32 output
// ============================================================
__global__ __launch_bounds__(256) void moe_gemm2(
    const unsigned short* __restrict__ hbuf, const float* __restrict__ W3,
    float* __restrict__ outbuf) {
  const int e  = blockIdx.z;
  const int mB = blockIdx.y * 128;
  const int nB = blockIdx.x * 64;

  __shared__ __align__(16) unsigned short lA[128 * 32];
  __shared__ __align__(16) unsigned short lB[ 64 * 32];

  const int tid  = threadIdx.x;
  const int lane = tid & 31;
  const int w    = tid >> 5;
  const int wm   = (w & 3) * 32;
  const int wn   = (w >> 2) * 32;

  floatx8 acc[2][2];
  #pragma unroll
  for (int i = 0; i < 2; ++i)
    #pragma unroll
    for (int j = 0; j < 2; ++j) acc[i][j] = zero8();

  const float* pW = W3 + (size_t)e * HH * DD;
  const unsigned short* pA = hbuf + (size_t)(e * CC + mB) * HH;
  const unsigned ldsA = (unsigned)(size_t)(void*)lA;

  const int rowL = lane & 15;
  const int c0   = (lane < 16) ? 0 : 8;
  const int kcol = (lane < 16) ? 0 : 8;
  const int nn4  = (tid & 15) * 4;
  const int kk2  = (tid >> 4) * 2;

  for (int k0 = 0; k0 < HH; k0 += 32) {
#if HAVE_TDM
    if (w == 0) tdm_load_2d(ldsA, pA + k0, 32, 128, HH);   // async A tile
#else
    {
      const int r = tid >> 1, seg = (tid & 1) * 16;
      const ux4* src = (const ux4*)(pA + (size_t)r * HH + k0 + seg);
      ux4* dst = (ux4*)(lA + r * 32 + seg);
      dst[0] = src[0];
      dst[1] = src[1];
    }
#endif
    {
      const float* r0 = pW + (size_t)(k0 + kk2) * DD + nB + nn4;
      if (k0 + 32 < HH) __builtin_prefetch(r0 + 32 * DD, 0, 1);
      fx4 a0 = *(const fx4*)r0;
      fx4 a1 = *(const fx4*)(r0 + DD);
      #pragma unroll
      for (int j = 0; j < 4; ++j)
        *(unsigned*)&lB[(nn4 + j) * 32 + kk2] =
            (unsigned)f2bf(a0[j]) | ((unsigned)f2bf(a1[j]) << 16);
    }
#if HAVE_TDM
    if (w == 0) __builtin_amdgcn_s_wait_tensorcnt(0);
#endif
    __syncthreads();

    bf16x16 aF[2], bF[2];
    #pragma unroll
    for (int mt = 0; mt < 2; ++mt) {
      const unsigned short* b = lA + (wm + mt * 16 + rowL) * 32 + c0;
      aF[mt] = ld_frag(b, b + 16);
    }
    #pragma unroll
    for (int nt = 0; nt < 2; ++nt) {
      const unsigned short* b = lB + (wn + nt * 16 + rowL) * 32 + kcol;
      bF[nt] = ld_frag(b, b + 16);
    }
    #pragma unroll
    for (int mt = 0; mt < 2; ++mt)
      #pragma unroll
      for (int nt = 0; nt < 2; ++nt)
        acc[mt][nt] = __builtin_amdgcn_wmma_f32_16x16x32_bf16(
            false, aF[mt], false, bF[nt], (short)0, acc[mt][nt], false, false);
    __syncthreads();
  }

  float* pO = outbuf + (size_t)(e * CC + mB) * DD + nB;
  const int nCol = lane & 15;
  const int rAdd = (lane < 16) ? 0 : 8;
  #pragma unroll
  for (int mt = 0; mt < 2; ++mt)
    #pragma unroll
    for (int nt = 0; nt < 2; ++nt)
      #pragma unroll
      for (int r = 0; r < 8; ++r) {
        const int m = wm + mt * 16 + rAdd + r;
        const int n = wn + nt * 16 + nCol;
        pO[(size_t)m * DD + n] = acc[mt][nt][r];
      }
}

// ============================================================
// 6) Combine: y[t] = w0*out[slot0] + w1*out[slot1]
// ============================================================
__global__ __launch_bounds__(256) void moe_combine(
    const float* __restrict__ outbuf, const int* __restrict__ slot_of,
    const float* __restrict__ topw, float* __restrict__ y) {
  const int t = blockIdx.x;
  const int d = threadIdx.x * 4;
  const int s0 = slot_of[t * 2 + 0];
  const int s1 = slot_of[t * 2 + 1];
  const float w0 = topw[t * 2 + 0];
  const float w1 = topw[t * 2 + 1];
  fx4 r = {0.f, 0.f, 0.f, 0.f};
  if (s0 >= 0) {
    fx4 a = *(const fx4*)(outbuf + (size_t)s0 * DD + d);
    r += w0 * a;
  }
  if (s1 >= 0) {
    fx4 a = *(const fx4*)(outbuf + (size_t)s1 * DD + d);
    r += w1 * a;
  }
  *(fx4*)(y + (size_t)t * DD + d) = r;
}

// ============================================================
extern "C" void kernel_launch(void* const* d_in, const int* in_sizes, int n_in,
                              void* d_out, int out_size, void* d_ws, size_t ws_size,
                              hipStream_t stream) {
  (void)in_sizes; (void)n_in; (void)out_size; (void)ws_size;
  const float* x  = (const float*)d_in[0];
  const float* Wg = (const float*)d_in[1];
  const float* bg = (const float*)d_in[2];
  const float* W1 = (const float*)d_in[3];
  const float* W2 = (const float*)d_in[4];
  const float* W3 = (const float*)d_in[5];
  float* y = (float*)d_out;

  char* p = (char*)d_ws;
  auto alloc = [&](size_t bytes) -> char* {
    char* r = p;
    p += (bytes + 255) & ~(size_t)255;
    return r;
  };
  int*            topi    = (int*)           alloc((size_t)TT * 2 * sizeof(int));
  float*          topw    = (float*)         alloc((size_t)TT * 2 * sizeof(float));
  int*            slot_of = (int*)           alloc((size_t)TT * 2 * sizeof(int));
  int*            tok_of  = (int*)           alloc((size_t)EE * CC * sizeof(int));
  unsigned short* disp    = (unsigned short*)alloc((size_t)EE * CC * DD * 2);
  unsigned short* hbuf    = (unsigned short*)alloc((size_t)EE * CC * HH * 2);
  float*          outbuf  = (float*)         alloc((size_t)EE * CC * DD * 4);

  moe_gate    <<<TT / 8, 256, 0, stream>>>(x, Wg, bg, topi, topw);
  moe_scan    <<<1,      256, 0, stream>>>(topi, slot_of, tok_of);
  moe_dispatch<<<EE * CC, 256, 0, stream>>>(x, tok_of, disp);

  dim3 g1(HH / 64, CC / 128, EE);   // 64 x 10 x 8
  moe_gemm1<<<g1, 256, 0, stream>>>(disp, W1, W2, hbuf);

  dim3 g2(DD / 64, CC / 128, EE);   // 16 x 10 x 8
  moe_gemm2<<<g2, 256, 0, stream>>>(hbuf, W3, outbuf);

  moe_combine<<<TT, 256, 0, stream>>>(outbuf, slot_of, topw, y);
}
